// AMPBlock0_77988016161215
// MI455X (gfx1250) — matrix operand order repaired and checked by reading, not compiled, not run on hardware
//
#include <hip/hip_runtime.h>

// ---------------------------------------------------------------------------
// AMPBlock0 for MI455X (gfx1250, wave32).
//   act -> conv(512,512,3) -> act -> conv -> act (+residual)
// Convs: bf16 WMMA GEMM (f32 accum), double-buffered LDS pipeline:
//   weights staged by the Tensor Data Mover (tensor_load_to_lds, LDS row
//   padding via D# pad fields), activations staged by vector loads; the
//   next tile's DMA overlaps the current tile's 8 WMMAs per wave.
// Acts: fused upFIR->snake->downFIR tile kernels (memory-bound path).
// ---------------------------------------------------------------------------

#define B_DIM 8
#define C_DIM 512
#define T_DIM 8192
#define KTOT  (3 * C_DIM)          // GEMM K = c_in * 3 = 1536
#define NKSTEP (KTOT / 32)         // 48
#define LDSW  40                   // padded LDS row: 32 elems + 8 pad (80 B)
#define NELEM ((size_t)B_DIM * C_DIM * T_DIM)

typedef __bf16 bf16;
typedef __attribute__((ext_vector_type(16))) __bf16 v16bf;
typedef __attribute__((ext_vector_type(8)))  float  v8f;
typedef __attribute__((ext_vector_type(4)))  unsigned int u32x4;
typedef __attribute__((ext_vector_type(8)))  int i32x8;
typedef __attribute__((ext_vector_type(4)))  int i32x4;

// Kaiser-sinc FIR, cutoff=0.25, half_width=0.3, KS=12 (beta=4.6638), normalized.
__constant__ float FILT[12] = {
     0.0020291f,  0.0093892f, -0.0255435f, -0.0576560f,
     0.1285745f,  0.4432066f,  0.4432066f,  0.1285745f,
    -0.0576560f, -0.0255435f,  0.0093892f,  0.0020291f };

// ---------------------------------------------------------------------------
// Fused alias-free activation: upsample(2x, FIR12) -> snake_beta -> down(2x).
// ---------------------------------------------------------------------------
template <typename IT, bool FINAL>
__global__ __launch_bounds__(256)
void act_kernel(const IT* __restrict__ in,
                const float* __restrict__ alpha_log,
                const float* __restrict__ beta_log,
                const float* __restrict__ resid,   // only for FINAL
                void* __restrict__ outv)
{
    constexpr int TN = 512;
    __shared__ float xs[TN + 16];        // x tile with +-8 halo
    __shared__ float vs[2 * TN + 16];    // snake(upsample) with +-8 halo

    const int t0 = blockIdx.x * TN;
    const int c  = blockIdx.y;
    const int b  = blockIdx.z;
    const size_t row = ((size_t)b * C_DIM + c) * T_DIM;

    const float a    = __expf(alpha_log[c]);
    const float invb = 1.0f / (__expf(beta_log[c]) + 1e-9f);

    for (int i = threadIdx.x; i < TN + 16; i += 256) {
        int g = t0 - 8 + i;
        g = min(max(g, 0), T_DIM - 1);
        xs[i] = (float)in[row + g];
    }
    __syncthreads();

    const int sbase = 2 * t0 - 8;
    for (int i = threadIdx.x; i < 2 * TN + 16; i += 256) {
        int s = sbase + i;
        s = min(max(s, 0), 2 * T_DIM - 1);
        const int tp  = s >> 1;
        const int par = s & 1;
        float u = 0.0f;
#pragma unroll
        for (int p = 0; p < 6; ++p) {
            const int xi = tp + p - 3 + par - (t0 - 8);
            u += FILT[2 * p + par] * xs[xi];
        }
        u *= 2.0f;                       // upsample gain r
        const float sn = __sinf(a * u);
        vs[i] = u + invb * sn * sn;      // snake_beta
    }
    __syncthreads();

    for (int tl = threadIdx.x; tl < TN; tl += 256) {
        float acc = 0.0f;
#pragma unroll
        for (int m = 0; m < 12; ++m) {
            int s = 2 * (t0 + tl) + m - 5;
            s = min(max(s, 0), 2 * T_DIM - 1);
            acc += FILT[m] * vs[s - sbase];
        }
        const size_t gi = row + t0 + tl;
        if (FINAL) ((float*)outv)[gi] = acc + resid[gi];
        else       ((bf16*)outv)[gi]  = (bf16)acc;
    }
}

// ---------------------------------------------------------------------------
// Repack fp32 weights [co][ci][k] -> bf16 [co][k*512+ci].
// ---------------------------------------------------------------------------
__global__ __launch_bounds__(256)
void wcvt_kernel(const float* __restrict__ w, bf16* __restrict__ wbf)
{
    const int idx = blockIdx.x * 256 + threadIdx.x;
    if (idx >= C_DIM * KTOT) return;
    const int co  = idx / KTOT;
    const int rem = idx - co * KTOT;
    const int k   = rem / C_DIM;
    const int ci  = rem - k * C_DIM;
    wbf[idx] = (bf16)w[((size_t)co * C_DIM + ci) * 3 + k];
}

// ---------------------------------------------------------------------------
// conv1d(k=3, pad=1, zero) as WMMA GEMM.
//  Block: 256 thr (8 waves). Tile: 128(co) x 128(t). K step 32, double-buffered.
//  Wave grid 4(M) x 2(N): each wave owns 32(co) x 64(t) = 2x4 WMMA tiles
//  -> 8 v_wmma_f32_16x16x32_bf16 per K-step from 2 A-frags / 4 B-frags,
//  each fragment = two ds_load_b128 (layout-contiguous), rows padded to 80 B.
// ---------------------------------------------------------------------------
__global__ __launch_bounds__(256)
void conv_wmma_kernel(const bf16* __restrict__ in,
                      const bf16* __restrict__ wts,
                      const float* __restrict__ bias,
                      bf16* __restrict__ out)
{
    __shared__ bf16 As[2][128][LDSW];   // [buf][co_local][k_local] (padded rows)
    __shared__ bf16 Bs[2][128][LDSW];   // [buf][t_local][k_local]

    const int tid  = threadIdx.x;
    const int t0   = blockIdx.x * 128;
    const int co0  = blockIdx.y * 128;
    const int b    = blockIdx.z;
    const int l    = tid & 31;     // lane in wave32
    const int wm   = tid >> 5;     // wave id 0..7
    const int wm2  = wm & 3;       // M sub-block (32 co)
    const int wn2  = wm >> 2;      // N sub-block (64 t)
    const int hf   = l >> 4;       // lane half
    const int ln   = l & 15;

    v8f acc[2][4];
#pragma unroll
    for (int mi = 0; mi < 2; ++mi)
#pragma unroll
        for (int j = 0; j < 4; ++j)
#pragma unroll
            for (int i = 0; i < 8; ++i) acc[mi][j][i] = 0.0f;

    // ---- stage K-tile kk into LDS buffer pb -------------------------------
    auto stage = [&](int kk, int pb) {
        // A (weights) tile: 32(K) x 128(co) via Tensor Data Mover.
        // D#: 2-byte elems, tensor 1536x512 stride 1536, tile 32x128,
        //     LDS pad: every 16 dwords (64 B row) insert 4 dwords (16 B).
        if (tid < 32) {
            const unsigned ldsA = (unsigned)(size_t)(&As[pb][0][0]);
            const unsigned long long ga =
                (unsigned long long)(const char*)(wts + (size_t)co0 * KTOT + kk);
            u32x4 g0;
            g0[0] = 0x1u;                                  // count=1, user mode
            g0[1] = ldsA;                                  // lds_addr (bytes)
            g0[2] = (unsigned)ga;                          // global_addr[31:0]
            g0[3] = (unsigned)(ga >> 32) | 0x80000000u;    // addr[56:32] | type=2
            i32x8 g1;
            g1[0] = (int)(0x00010000u                      // data_size=1 (2B)
                          | (1u << 20)                     // pad_enable
                          | (3u << 22)                     // pad_interval: 16 dw
                          | (3u << 25));                   // pad_amount: 4 dw
            g1[1] = (int)((KTOT & 0xFFFFu) << 16);         // tensor_dim0 lo16
            g1[2] = (int)(((unsigned)KTOT >> 16) |
                          ((C_DIM & 0xFFFFu) << 16));      // dim0 hi | dim1 lo
            g1[3] = (int)(((unsigned)C_DIM >> 16) |
                          (32u << 16));                    // dim1 hi | tile_dim0=32
            g1[4] = 128;                                   // tile_dim1=128 rows
            g1[5] = KTOT;                                  // tensor_dim0_stride
            g1[6] = 0;
            g1[7] = 0;
            const i32x4 gz4 = {0, 0, 0, 0};
            const i32x8 gz8 = {0, 0, 0, 0, 0, 0, 0, 0};
            __builtin_amdgcn_tensor_load_to_lds(g0, g1, gz4, gz4, gz8, 0);
        }
        // B (activation) tile: in[b][ci0+r][t0+tt-1+k], zero-padded t edges.
        {
            const int tt = tid & 127;
            const int rb = (tid >> 7) * 16;
            const int k   = kk >> 9;        // kk / 512
            const int ci0 = kk & 511;       // kk % 512
            const int tg  = t0 + tt - 1 + k;
            union { uint4 v[2]; bf16 h[16]; } tmp;
#pragma unroll
            for (int q = 0; q < 16; ++q) {
                bf16 val = (bf16)0.0f;
                if (tg >= 0 && tg < T_DIM)
                    val = in[((size_t)b * C_DIM + ci0 + rb + q) * T_DIM + tg];
                tmp.h[q] = val;
            }
            uint4* dst = (uint4*)&Bs[pb][tt][rb];
            dst[0] = tmp.v[0];
            dst[1] = tmp.v[1];
        }
    };

    // ---- 8 WMMAs on LDS buffer pb -----------------------------------------
    auto compute = [&](int pb) {
        union Frag { v16bf v; uint4 q[2]; };
        Frag af[2];
#pragma unroll
        for (int mi = 0; mi < 2; ++mi) {
            const int rowA = wm2 * 32 + mi * 16 + ln;
            // A 16x32 layout: per-lane dwords are contiguous -> 2x b128
            af[mi].q[0] = *(const uint4*)&As[pb][rowA][hf * 8];
            af[mi].q[1] = *(const uint4*)&As[pb][rowA][16 + hf * 8];
        }
#pragma unroll
        for (int j = 0; j < 4; ++j) {
            Frag bfr;
            const int tl2 = wn2 * 64 + j * 16 + ln;
            // B 32x16 layout: lanes<16 K0-7,16-23, lanes>=16 K8-15,24-31 -> 2x b128
            bfr.q[0] = *(const uint4*)&Bs[pb][tl2][hf * 8];
            bfr.q[1] = *(const uint4*)&Bs[pb][tl2][16 + hf * 8];
#pragma unroll
            for (int mi = 0; mi < 2; ++mi)
                acc[mi][j] = __builtin_amdgcn_wmma_f32_16x16x32_bf16(
                    false, af[mi].v, false, bfr.v, (short)0, acc[mi][j],
                    false, false);
        }
    };

    // ---- software pipeline: stage(i+1) overlaps compute(i) ----------------
    stage(0, 0);
    if (tid < 32) __builtin_amdgcn_s_wait_tensorcnt(0);
    __syncthreads();

    for (int it = 0; it < NKSTEP; ++it) {
        const int pb = it & 1;
        if (it + 1 < NKSTEP) stage((it + 1) * 32, pb ^ 1);
        compute(pb);
        if (tid < 32) __builtin_amdgcn_s_wait_tensorcnt(0);
        __syncthreads();
    }

    // ---- epilogue: C/D layout — VGPR i: M = hf*8 + i, N = ln
#pragma unroll
    for (int mi = 0; mi < 2; ++mi) {
#pragma unroll
        for (int j = 0; j < 4; ++j) {
#pragma unroll
            for (int i = 0; i < 8; ++i) {
                const int co = co0 + wm2 * 32 + mi * 16 + hf * 8 + i;
                const int t  = t0 + wn2 * 64 + j * 16 + ln;
                out[((size_t)b * C_DIM + co) * T_DIM + t] = (bf16)(acc[mi][j][i] + bias[co]);
            }
        }
    }
}

// ---------------------------------------------------------------------------
// Launch: wcvt x2, act1, conv1, act2, conv2, act_final(+residual).
// ws layout: bufA(bf16 64MB) | bufB(bf16 64MB) | wbf1 | wbf2  (~137.5 MB)
// ---------------------------------------------------------------------------
extern "C" void kernel_launch(void* const* d_in, const int* in_sizes, int n_in,
                              void* d_out, int out_size, void* d_ws, size_t ws_size,
                              hipStream_t stream)
{
    const float* x   = (const float*)d_in[0];
    const float* w1  = (const float*)d_in[1];
    const float* b1  = (const float*)d_in[2];
    const float* w2  = (const float*)d_in[3];
    const float* b2  = (const float*)d_in[4];
    const float* al1 = (const float*)d_in[5];
    const float* be1 = (const float*)d_in[6];
    const float* al2 = (const float*)d_in[7];
    const float* be2 = (const float*)d_in[8];

    char* ws   = (char*)d_ws;
    bf16* bufA = (bf16*)ws;
    bf16* bufB = (bf16*)(ws + NELEM * 2);
    bf16* wbf1 = (bf16*)(ws + NELEM * 4);
    bf16* wbf2 = wbf1 + (size_t)C_DIM * KTOT;

    const int nw = C_DIM * KTOT;                      // 786432
    wcvt_kernel<<<(nw + 255) / 256, 256, 0, stream>>>(w1, wbf1);
    wcvt_kernel<<<(nw + 255) / 256, 256, 0, stream>>>(w2, wbf2);

    dim3 ga(T_DIM / 512, C_DIM, B_DIM);               // act tiles
    dim3 gc(T_DIM / 128, C_DIM / 128, B_DIM);         // conv tiles

    act_kernel<float, false><<<ga, 256, 0, stream>>>(x, al1, be1, nullptr, bufA);
    conv_wmma_kernel<<<gc, 256, 0, stream>>>(bufA, wbf1, b1, bufB);
    act_kernel<bf16, false><<<ga, 256, 0, stream>>>(bufB, al1, be1, nullptr, bufA);
    conv_wmma_kernel<<<gc, 256, 0, stream>>>(bufA, wbf2, b2, bufB);
    act_kernel<bf16, true><<<ga, 256, 0, stream>>>(bufB, al2, be2, x, d_out);
}
